// Attention_927712936452
// MI455X (gfx1250) — matrix operand order repaired and checked by reading, not compiled, not run on hardware
//
#include <hip/hip_runtime.h>
#include <hip/hip_bf16.h>

#define DEV __device__ __forceinline__

typedef __attribute__((ext_vector_type(16))) __bf16 v16bf;
typedef __attribute__((ext_vector_type(8)))  __bf16 v8bf;
typedef __attribute__((ext_vector_type(8)))  float  v8f;
typedef __attribute__((ext_vector_type(4)))  int    v4i;

namespace {
constexpr int B_  = 8;
constexpr int N_  = 1024;
constexpr int L_  = 1024;
constexpr int S_  = 2048;    // N_ + L_
constexpr int C_  = 768;
constexpr int H_  = 12;
constexpr int DH_ = 64;
constexpr int C3_ = 3 * C_;  // 2304
constexpr int M_  = B_ * S_; // 16384 rows in the big GEMMs
}

// Async global->LDS staging path (CDNA5 GLOBAL_LOAD_ASYNC_TO_LDS_B128, ASYNCcnt).
#if defined(__has_builtin)
#if __has_builtin(__builtin_amdgcn_global_load_async_to_lds_b128) && \
    __has_builtin(__builtin_amdgcn_s_wait_asynccnt)
#define ATTN_ASYNC 1
#endif
#endif
#ifndef ATTN_ASYNC
#define ATTN_ASYNC 0
#endif

// ---------------- helpers ----------------

DEV __bf16 f2bf(float f) {
    unsigned u = __builtin_bit_cast(unsigned, f);
    unsigned r = u + 0x7FFFu + ((u >> 16) & 1u);   // round-to-nearest-even
    unsigned short s = (unsigned short)(r >> 16);
    return __builtin_bit_cast(__bf16, s);
}

// A fragment (16x32 bf16, M x K). Row m = lane%16.
// lanes 0-15 : elems 0..7 -> K 0..7,   elems 8..15 -> K 16..23
// lanes 16-31: elems 0..7 -> K 8..15,  elems 8..15 -> K 24..31
DEV v16bf load_frag_a(const __bf16* base, int ld) {
    int lane = threadIdx.x & 31;
    int m   = lane & 15;
    int hi  = lane >> 4;
    const __bf16* p = base + (size_t)m * ld + hi * 8;
    union { v16bf v; v8bf h[2]; } u;
    u.h[0] = *(const v8bf*)(p);
    u.h[1] = *(const v8bf*)(p + 16);
    return u.v;
}

// B fragment (32x16 bf16, K x N). Column n = lane%16 (contiguous over K in memory:
// "base" points at a row-major [N,K] matrix whose rows are the B columns).
// lanes 0-15: elems 0..7 -> K 0..7, 8..15 -> K 16..23 ; lanes 16-31: K 8..15, K 24..31
DEV v16bf load_frag_b(const __bf16* base, int ld) {
    int lane = threadIdx.x & 31;
    int n   = lane & 15;
    int hi  = lane >> 4;
    const __bf16* p = base + (size_t)n * ld + hi * 8;
    union { v16bf v; v8bf h[2]; } u;
    u.h[0] = *(const v8bf*)(p);
    u.h[1] = *(const v8bf*)(p + 16);
    return u.v;
}

DEV v8f wmma_bf(v16bf a, v16bf b, v8f c) {
    return __builtin_amdgcn_wmma_f32_16x16x32_bf16(false, a, false, b, (short)0, c,
                                                   false, false);
}

DEV float half16_max(float v) {
#pragma unroll
    for (int m = 1; m < 16; m <<= 1) v = fmaxf(v, __shfl_xor(v, m, 32));
    return v;
}
DEV float half16_sum(float v) {
#pragma unroll
    for (int m = 1; m < 16; m <<= 1) v += __shfl_xor(v, m, 32);
    return v;
}

#if ATTN_ASYNC
typedef __attribute__((address_space(1))) v4i* gv4i_ptr;
typedef __attribute__((address_space(3))) v4i* lv4i_ptr;
// one 16B async global->LDS transfer per lane
DEV void async_copy16(const void* g, void* l) {
    __builtin_amdgcn_global_load_async_to_lds_b128(
        (gv4i_ptr)(unsigned long long)g,
        (lv4i_ptr)(unsigned int)(unsigned long long)l, 0, 0);
}
#endif

// ---------------- kernel 0: fp32 -> bf16 converts ----------------

__global__ __launch_bounds__(256) void convert_z_kernel(const float* __restrict__ x,
                                                        const float* __restrict__ y,
                                                        __bf16* __restrict__ zb) {
    size_t i = ((size_t)blockIdx.x * blockDim.x + threadIdx.x) * 4;
    int    c   = (int)(i % C_);
    size_t row = i / C_;                 // b*S_ + s
    int    s   = (int)(row & (S_ - 1));
    int    b   = (int)(row >> 11);
    const float* src = (s < N_) ? (x + ((size_t)b * N_ + s) * C_ + c)
                                : (y + ((size_t)b * L_ + (s - N_)) * C_ + c);
    float4 f = *(const float4*)src;
    zb[i + 0] = f2bf(f.x); zb[i + 1] = f2bf(f.y);
    zb[i + 2] = f2bf(f.z); zb[i + 3] = f2bf(f.w);
}

__global__ __launch_bounds__(256) void convert_w_kernel(const float* __restrict__ w,
                                                        __bf16* __restrict__ wb, int n4) {
    int i4 = blockIdx.x * blockDim.x + threadIdx.x;
    if (i4 >= n4) return;
    int i = i4 * 4;
    float4 f = *(const float4*)(w + i);
    wb[i + 0] = f2bf(f.x); wb[i + 1] = f2bf(f.y);
    wb[i + 2] = f2bf(f.z); wb[i + 3] = f2bf(f.w);
}

// ---------------- kernel 1: QKV GEMM + scatter ----------------
// [M_,C_] x [C3_,C_]^T -> q[B,H,S,Dh] (scaled), k[B,H,S,Dh], vT[B,H,Dh,S]
// block = 256 thr (8 waves, 4Mx2N), block tile 128x128, wave tile 32x64.

__global__ __launch_bounds__(256) void qkv_gemm_kernel(const __bf16* __restrict__ zb,
                                                       const __bf16* __restrict__ wq,
                                                       __bf16* __restrict__ qb,
                                                       __bf16* __restrict__ kbuf,
                                                       __bf16* __restrict__ vtb) {
    int lane = threadIdx.x & 31;
    int wave = threadIdx.x >> 5;
    int wm = wave & 3, wn = wave >> 2;
    int m0 = blockIdx.x * 128 + wm * 32;
    int n0 = blockIdx.y * 128 + wn * 64;

    v8f acc[2][4];
#pragma unroll
    for (int i = 0; i < 2; i++)
#pragma unroll
        for (int j = 0; j < 4; j++)
#pragma unroll
            for (int r = 0; r < 8; r++) acc[i][j][r] = 0.f;

    for (int kk = 0; kk < C_; kk += 32) {
        v16bf a0 = load_frag_a(zb + (size_t)m0 * C_ + kk, C_);
        v16bf a1 = load_frag_a(zb + (size_t)(m0 + 16) * C_ + kk, C_);
        v16bf b0 = load_frag_b(wq + (size_t)(n0 +  0) * C_ + kk, C_);
        v16bf b1 = load_frag_b(wq + (size_t)(n0 + 16) * C_ + kk, C_);
        v16bf b2 = load_frag_b(wq + (size_t)(n0 + 32) * C_ + kk, C_);
        v16bf b3 = load_frag_b(wq + (size_t)(n0 + 48) * C_ + kk, C_);
        acc[0][0] = wmma_bf(a0, b0, acc[0][0]);
        acc[0][1] = wmma_bf(a0, b1, acc[0][1]);
        acc[0][2] = wmma_bf(a0, b2, acc[0][2]);
        acc[0][3] = wmma_bf(a0, b3, acc[0][3]);
        acc[1][0] = wmma_bf(a1, b0, acc[1][0]);
        acc[1][1] = wmma_bf(a1, b1, acc[1][1]);
        acc[1][2] = wmma_bf(a1, b2, acc[1][2]);
        acc[1][3] = wmma_bf(a1, b3, acc[1][3]);
    }

    int half8 = (lane >> 4) << 3;
    int col   = lane & 15;
#pragma unroll
    for (int i = 0; i < 2; i++) {
#pragma unroll
        for (int j = 0; j < 4; j++) {
            int d0  = n0 + j * 16;         // 16-wide tile never crosses a 64 (head) bdy
            int t   = d0 / C_;             // 0=q 1=k 2=v
            int hd  = d0 - t * C_;
            int h   = hd >> 6;
            int dh  = (hd & 63) + col;
#pragma unroll
            for (int r = 0; r < 8; r++) {
                int g = m0 + i * 16 + r + half8;     // global row in [0, M_)
                int b = g >> 11;
                int s = g & (S_ - 1);
                float v = acc[i][j][r];
                if (t == 0) {
                    qb[(((size_t)b * H_ + h) * S_ + s) * DH_ + dh] = f2bf(v * 0.125f);
                } else if (t == 1) {
                    kbuf[(((size_t)b * H_ + h) * S_ + s) * DH_ + dh] = f2bf(v);
                } else {
                    vtb[(((size_t)b * H_ + h) * DH_ + dh) * S_ + s] = f2bf(v);
                }
            }
        }
    }
}

// ---------------- kernel 2: flash attention ----------------
// grid (32 query-tiles, B*H). block = 128 thr (4 waves), wave owns 16 queries.
// K/V 64x64 tiles are staged in LDS via async global->LDS (when available),
// removing the 4x per-wave duplicate global fragment loads.

__global__ __launch_bounds__(128) void attn_kernel(const __bf16* __restrict__ qb,
                                                   const __bf16* __restrict__ kbuf,
                                                   const __bf16* __restrict__ vtb,
                                                   __bf16* __restrict__ attn_out) {
    int qt   = blockIdx.x;             // 0..31
    int bh   = blockIdx.y;             // 0..95
    int wave = threadIdx.x >> 5;
    int lane = threadIdx.x & 31;
    int half8 = (lane >> 4) << 3;
    int col   = lane & 15;

    int q0     = qt * 64 + wave * 16;
    int kv_len = (qt * 64 < N_) ? N_ : S_;   // x-queries: self only; y-queries: full

    const __bf16* qbase = qb   + (size_t)bh * S_ * DH_ + (size_t)q0 * DH_;
    const __bf16* kbase = kbuf + (size_t)bh * S_ * DH_;
    const __bf16* vbase = vtb  + (size_t)bh * DH_ * S_;

    v16bf aq0 = load_frag_a(qbase +  0, DH_);
    v16bf aq1 = load_frag_a(qbase + 32, DH_);

    __shared__ __bf16 lds_p[4][16][64];   // wave-private 16x64 P tile (bf16)
#if ATTN_ASYNC
    __shared__ __bf16 k_tile[64][64];     // [key][dh]
    __shared__ __bf16 v_tile[64][64];     // [dh][key]
#endif

    float mrow[8], lrow[8];
    v8f o[4];
#pragma unroll
    for (int r = 0; r < 8; r++) { mrow[r] = -1e30f; lrow[r] = 0.f; }
#pragma unroll
    for (int n = 0; n < 4; n++)
#pragma unroll
        for (int r = 0; r < 8; r++) o[n][r] = 0.f;

    for (int k0 = 0; k0 < kv_len; k0 += 64) {
#if ATTN_ASYNC
        // cooperative async staging: 2 tiles x 8KB = 512+512 x 16B transfers
#pragma unroll
        for (int c = 0; c < 4; c++) {
            int chunk = c * 128 + (int)threadIdx.x;   // 0..511, lane-consecutive
            int row   = chunk >> 3;                   // 0..63
            int off8  = (chunk & 7) * 8;              // bf16 element offset
            async_copy16(kbase + (size_t)(k0 + row) * DH_ + off8, &k_tile[row][off8]);
            async_copy16(vbase + (size_t)row * S_ + k0 + off8,    &v_tile[row][off8]);
        }
        __builtin_amdgcn_s_wait_asynccnt(0);
        __syncthreads();
#endif
        // scores: 16 queries x 64 keys
        v8f sc[4];
#pragma unroll
        for (int j = 0; j < 4; j++) {
#if ATTN_ASYNC
            const __bf16* kr = &k_tile[j * 16][0];
            v16bf bk0 = load_frag_b(kr +  0, 64);
            v16bf bk1 = load_frag_b(kr + 32, 64);
#else
            const __bf16* kr = kbase + (size_t)(k0 + j * 16) * DH_;
            v16bf bk0 = load_frag_b(kr +  0, DH_);
            v16bf bk1 = load_frag_b(kr + 32, DH_);
#endif
            v8f z;
#pragma unroll
            for (int r = 0; r < 8; r++) z[r] = 0.f;
            sc[j] = wmma_bf(aq0, bk0, z);
            sc[j] = wmma_bf(aq1, bk1, sc[j]);
        }
        // online softmax per row (row = r + half8, spread across a 16-lane half)
#pragma unroll
        for (int r = 0; r < 8; r++) {
            float mc = fmaxf(fmaxf(sc[0][r], sc[1][r]), fmaxf(sc[2][r], sc[3][r]));
            mc = half16_max(mc);
            float mn    = fmaxf(mrow[r], mc);
            float alpha = __expf(mrow[r] - mn);
            mrow[r] = mn;
            float rs = 0.f;
#pragma unroll
            for (int j = 0; j < 4; j++) {
                float p = __expf(sc[j][r] - mn);
                sc[j][r] = p;
                rs += p;
            }
            rs = half16_sum(rs);
            lrow[r] = lrow[r] * alpha + rs;
#pragma unroll
            for (int n = 0; n < 4; n++) o[n][r] *= alpha;
            int row = r + half8;
#pragma unroll
            for (int j = 0; j < 4; j++)
                lds_p[wave][row][j * 16 + col] = f2bf(sc[j][r]);
        }
        // C-layout -> A-layout through LDS (wave-private, DS ordering suffices)
        const __bf16* pl = &lds_p[wave][0][0];
        v16bf ap0 = load_frag_a(pl +  0, 64);
        v16bf ap1 = load_frag_a(pl + 32, 64);
        // O += P @ V   (N = Dh, B columns are rows of vT -> contiguous over keys)
#pragma unroll
        for (int n = 0; n < 4; n++) {
#if ATTN_ASYNC
            const __bf16* vr = &v_tile[n * 16][0];
            v16bf bv0 = load_frag_b(vr +  0, 64);
            v16bf bv1 = load_frag_b(vr + 32, 64);
#else
            const __bf16* vr = vbase + (size_t)(n * 16) * S_ + k0;
            v16bf bv0 = load_frag_b(vr +  0, S_);
            v16bf bv1 = load_frag_b(vr + 32, S_);
#endif
            o[n] = wmma_bf(ap0, bv0, o[n]);
            o[n] = wmma_bf(ap1, bv1, o[n]);
        }
#if ATTN_ASYNC
        __syncthreads();   // all waves done reading tiles before next overwrite
#endif
    }

    int b = bh / H_, h = bh - b * H_;
#pragma unroll
    for (int r = 0; r < 8; r++) {
        float inv = 1.0f / lrow[r];
        int row = q0 + r + half8;
#pragma unroll
        for (int n = 0; n < 4; n++) {
            attn_out[((size_t)b * S_ + row) * C_ + h * DH_ + n * 16 + col] =
                f2bf(o[n][r] * inv);
        }
    }
}

// ---------------- kernel 3: proj GEMM + bias + output split ----------------

__global__ __launch_bounds__(256) void proj_gemm_kernel(const __bf16* __restrict__ attn,
                                                        const __bf16* __restrict__ wp,
                                                        const float* __restrict__ bias,
                                                        float* __restrict__ out) {
    int lane = threadIdx.x & 31;
    int wave = threadIdx.x >> 5;
    int wm = wave & 3, wn = wave >> 2;
    int m0 = blockIdx.x * 128 + wm * 32;
    int n0 = blockIdx.y * 128 + wn * 64;

    v8f acc[2][4];
#pragma unroll
    for (int i = 0; i < 2; i++)
#pragma unroll
        for (int j = 0; j < 4; j++)
#pragma unroll
            for (int r = 0; r < 8; r++) acc[i][j][r] = 0.f;

    for (int kk = 0; kk < C_; kk += 32) {
        v16bf a0 = load_frag_a(attn + (size_t)m0 * C_ + kk, C_);
        v16bf a1 = load_frag_a(attn + (size_t)(m0 + 16) * C_ + kk, C_);
        v16bf b0 = load_frag_b(wp + (size_t)(n0 +  0) * C_ + kk, C_);
        v16bf b1 = load_frag_b(wp + (size_t)(n0 + 16) * C_ + kk, C_);
        v16bf b2 = load_frag_b(wp + (size_t)(n0 + 32) * C_ + kk, C_);
        v16bf b3 = load_frag_b(wp + (size_t)(n0 + 48) * C_ + kk, C_);
        acc[0][0] = wmma_bf(a0, b0, acc[0][0]);
        acc[0][1] = wmma_bf(a0, b1, acc[0][1]);
        acc[0][2] = wmma_bf(a0, b2, acc[0][2]);
        acc[0][3] = wmma_bf(a0, b3, acc[0][3]);
        acc[1][0] = wmma_bf(a1, b0, acc[1][0]);
        acc[1][1] = wmma_bf(a1, b1, acc[1][1]);
        acc[1][2] = wmma_bf(a1, b2, acc[1][2]);
        acc[1][3] = wmma_bf(a1, b3, acc[1][3]);
    }

    int half8 = (lane >> 4) << 3;
    int col   = lane & 15;
#pragma unroll
    for (int i = 0; i < 2; i++) {
#pragma unroll
        for (int j = 0; j < 4; j++) {
            int d  = n0 + j * 16 + col;
            float bb = bias[d];
#pragma unroll
            for (int r = 0; r < 8; r++) {
                int g = m0 + i * 16 + r + half8;
                int b = g >> 11;
                int s = g & (S_ - 1);
                size_t dst = (s < N_)
                    ? ((size_t)b * N_ + s) * C_ + d                       // x_out block
                    : (size_t)B_ * N_ * C_ + ((size_t)b * L_ + (s - N_)) * C_ + d;
                out[dst] = acc[i][j][r] + bb;
            }
        }
    }
}

// ---------------- launch ----------------

extern "C" void kernel_launch(void* const* d_in, const int* in_sizes, int n_in,
                              void* d_out, int out_size, void* d_ws, size_t ws_size,
                              hipStream_t stream) {
    const float* x      = (const float*)d_in[0];
    const float* y      = (const float*)d_in[1];
    const float* qkv_w  = (const float*)d_in[2];
    const float* proj_w = (const float*)d_in[3];
    const float* proj_b = (const float*)d_in[4];
    float* out = (float*)d_out;

    char* ws = (char*)d_ws;
    size_t off = 0;
    auto take = [&](size_t bytes) -> char* {
        char* p = ws + off;
        off += (bytes + 255) & ~(size_t)255;
        return p;
    };
    __bf16* zb   = (__bf16*)take((size_t)M_ * C_ * 2);          // concat(x,y) bf16
    __bf16* wqb  = (__bf16*)take((size_t)C3_ * C_ * 2);         // qkv_w bf16
    __bf16* wpb  = (__bf16*)take((size_t)C_ * C_ * 2);          // proj_w bf16
    __bf16* qb   = (__bf16*)take((size_t)B_ * H_ * S_ * DH_ * 2);   // Q (scaled)
    __bf16* kbuf = (__bf16*)take((size_t)B_ * H_ * S_ * DH_ * 2);   // K
    __bf16* vtb  = (__bf16*)take((size_t)B_ * H_ * DH_ * S_ * 2);   // V^T
    __bf16* attn = (__bf16*)take((size_t)M_ * C_ * 2);          // attention output
    if (off > ws_size) return;   // workspace too small: nothing sane to do

    // converts
    {
        int blocks = (int)(((size_t)M_ * C_ / 4) / 256);        // 12288
        convert_z_kernel<<<blocks, 256, 0, stream>>>(x, y, zb);
    }
    {
        int n4 = C3_ * C_ / 4;
        convert_w_kernel<<<(n4 + 255) / 256, 256, 0, stream>>>(qkv_w, wqb, n4);
    }
    {
        int n4 = C_ * C_ / 4;
        convert_w_kernel<<<(n4 + 255) / 256, 256, 0, stream>>>(proj_w, wpb, n4);
    }
    // QKV GEMM: 16384 x 2304 x 768
    qkv_gemm_kernel<<<dim3(M_ / 128, C3_ / 128), 256, 0, stream>>>(zb, wqb, qb, kbuf, vtb);
    // attention: 32 query tiles x (B*H) heads
    attn_kernel<<<dim3(S_ / 64, B_ * H_), 128, 0, stream>>>(qb, kbuf, vtb, attn);
    // proj GEMM: 16384 x 768 x 768, writes split outputs
    proj_gemm_kernel<<<dim3(M_ / 128, C_ / 128), 256, 0, stream>>>(attn, wpb, proj_b, out);
}
